// NeuralFingerprint_56710748176713
// MI455X (gfx1250) — matrix operand order repaired and checked by reading, not compiled, not run on hardware
//
#include <hip/hip_runtime.h>
#include <hip/hip_bf16.h>
#include <stdint.h>
#include <stddef.h>

#if __has_include(<hip/amd_detail/amd_gfx1250_TDM.h>)
#define TDM_6ARG 1
#else
#define TDM_6ARG 0
#endif

#define N_ATOMS 131072
#define N_EDGES 262144
#define N_MOLS  4096
#define ND_G    32768
#define EDGE_F  64
#define BN_EPS  1e-5f
#define BN_GROUPS 512

typedef __bf16 bf16;
typedef __attribute__((ext_vector_type(16))) __bf16 v16bf;
typedef __attribute__((ext_vector_type(8)))  float  v8f;
typedef __attribute__((ext_vector_type(4)))  unsigned int u32x4;
typedef __attribute__((ext_vector_type(4)))  int i32x4;
typedef __attribute__((ext_vector_type(8)))  int i32x8;

union Frag128 { v16bf v; u32x4 u[2]; };

// ---------------------------------------------------------------- utilities

__global__ void k_zero_f32(float* __restrict__ p, long n) {
  long i = (long)blockIdx.x * blockDim.x + threadIdx.x;
  if (i < n) p[i] = 0.f;
}

__global__ void k_f32_to_bf16(const float* __restrict__ src, bf16* __restrict__ dst, long n) {
  long i = (long)blockIdx.x * blockDim.x + threadIdx.x;
  if (i < n) dst[i] = (bf16)src[i];
}

// W [K,Nc] f32 row-major  ->  WT [Nc,K] bf16 row-major
__global__ void k_transpose_bf16(const float* __restrict__ src, bf16* __restrict__ dst,
                                 int K, int Nc) {
  long i = (long)blockIdx.x * blockDim.x + threadIdx.x;
  long total = (long)K * Nc;
  if (i >= total) return;
  int k = (int)(i / Nc), n = (int)(i % Nc);
  dst[(long)n * K + k] = (bf16)src[i];
}

// ------------------------------------------------------------- WMMA GEMM
// C[M,Nc] (+)= A[M,K] @ W[K,Nc] + bias,  A bf16 row-major, W given as WT[Nc,K] bf16.
// Block tile 128x128 (8 waves, each wave 16(M) x 128(N)). B tile staged in LDS
// (row stride K+8 bf16 => 16B pad per row => bank-conflict-free ds_read_b128).
// B-tile fill via Tensor Data Mover when K/2 dwords is a power of two, else
// cooperative vectorized copy. A fragments (global) and B fragments (LDS) are
// both ping-pong double-buffered; K-loop unrolled x2 (all step counts even) so
// no register copies / WMMA->VALU hazard bubbles.
// Fragment layouts per CDNA5 ISA 7.12.2 (h = lane>>4):
//   A lane L(row m0+(L&15)): elems 0..7 = K k0+h*8.. , elems 8..15 = K k0+16+h*8..
//   B lane L(col n0+(L&15)): elems 0..7 = K k0+h*8.. , elems 8..15 = K k0+16+h*8..
//   C lane L, vgpr e: row m0 + h*8 + e, col n0+(L&15)

__device__ __forceinline__ void kstep8(v8f* acc, const Frag128& a,
                                       const bf16* bbase, int ldw, int k0) {
  Frag128 b0, b1;
  const bf16* bp0 = bbase + k0;
  b0.u[0] = *(const u32x4*)(bp0);
  b0.u[1] = *(const u32x4*)(bp0 + 16);
#pragma unroll
  for (int g = 0; g < 8; ++g) {
    Frag128& cur = (g & 1) ? b1 : b0;
    Frag128& nxt = (g & 1) ? b0 : b1;
    if (g < 7) {                               // prefetch next B fragment
      const bf16* bp = bbase + (size_t)(g + 1) * 16 * ldw + k0;
      nxt.u[0] = *(const u32x4*)(bp);
      nxt.u[1] = *(const u32x4*)(bp + 16);
    }
    acc[g] = __builtin_amdgcn_wmma_f32_16x16x32_bf16(
        false, a.v, false, cur.v, (short)0, acc[g], false, false);
  }
}

__global__ __launch_bounds__(256)
void k_gemm_bf16_wmma(const bf16* __restrict__ A, const bf16* __restrict__ WT,
                      const float* __restrict__ bias, float* __restrict__ C,
                      int M, int K, int Nc, int accumulate, int use_tdm) {
  extern __shared__ bf16 smem[];                 // [128][K+8]
  const int tid  = threadIdx.x;
  const int lane = tid & 31;
  const int wave = tid >> 5;
  const int n0 = blockIdx.x * 128;
  const int m0 = blockIdx.y * 128 + wave * 16;
  const int ldw = K + 8;

  // ---- stage B tile (rows n0..n0+127 of WT, full K) into LDS ----
  if (use_tdm) {
    if (wave == 0) {
      unsigned long long ga = (unsigned long long)(uintptr_t)(WT + (size_t)n0 * K);
      unsigned lds0 = (unsigned)(uintptr_t)smem;
      int padI = (K == 128) ? 5 : (K == 256) ? 6 : 7;  // pad interval = K/2 dwords
      u32x4 g0;
      g0[0] = 1u;                                       // count=1, user descriptor
      g0[1] = lds0;                                     // lds_addr
      g0[2] = (unsigned)(ga & 0xffffffffu);             // global_addr lo
      g0[3] = (unsigned)((ga >> 32) & 0x1ffffffu) | (2u << 30);  // addr hi | type=2
      i32x8 g1;
      g1[0] = (1 << 16) | (1 << 20) | (padI << 22) | (3 << 25); // 2B elems, pad_en, pad=4dw
      g1[1] = (K & 0xffff) << 16;        // tensor_dim0[15:0]
      g1[2] = (Nc & 0xffff) << 16;       // tensor_dim0 hi=0 | tensor_dim1[15:0]
      g1[3] = (K & 0xffff) << 16;        // tensor_dim1 hi=0 | tile_dim0 = K
      g1[4] = 128;                       // tile_dim1 = 128, tile_dim2 = 0
      g1[5] = K;                         // tensor_dim0_stride lo
      g1[6] = 0;
      g1[7] = 0;
      i32x4 z4 = {0, 0, 0, 0};
#if TDM_6ARG
      i32x8 z8 = {0, 0, 0, 0, 0, 0, 0, 0};
      __builtin_amdgcn_tensor_load_to_lds(g0, g1, z4, z4, z8, 0);
#else
      __builtin_amdgcn_tensor_load_to_lds(g0, g1, z4, z4, 0);
#endif
      __builtin_amdgcn_s_wait_tensorcnt(0);
    }
    __syncthreads();
  } else {
    const int kv = K / 8;                       // 16B vectors per row
    for (int v = tid; v < 128 * kv; v += 256) {
      int row = v / kv;
      int k8  = (v % kv) * 8;
      *(u32x4*)(smem + (size_t)row * ldw + k8) =
          *(const u32x4*)(WT + (size_t)(n0 + row) * K + k8);
    }
    __syncthreads();
  }

  const int half = lane >> 4;
  const int l16  = lane & 15;

  v8f acc[8];
#pragma unroll
  for (int g = 0; g < 8; ++g)
#pragma unroll
    for (int e = 0; e < 8; ++e) acc[g][e] = 0.f;

  const bf16* arp   = A + (size_t)(m0 + l16) * K + half * 8;
  const bf16* bbase = smem + (size_t)l16 * ldw + half * 8;

  // A fragment ping-pong: aA covers k0, aB covers k0+32; reload only after use.
  Frag128 aA, aB;
  aA.u[0] = *(const u32x4*)(arp);
  aA.u[1] = *(const u32x4*)(arp + 16);
  aB.u[0] = *(const u32x4*)(arp + 32);
  aB.u[1] = *(const u32x4*)(arp + 48);
  for (int k0 = 0; k0 < K; k0 += 64) {
    kstep8(acc, aA, bbase, ldw, k0);
    if (k0 + 64 < K) {
      aA.u[0] = *(const u32x4*)(arp + k0 + 64);
      aA.u[1] = *(const u32x4*)(arp + k0 + 80);
    }
    kstep8(acc, aB, bbase, ldw, k0 + 32);
    if (k0 + 96 < K) {
      aB.u[0] = *(const u32x4*)(arp + k0 + 96);
      aB.u[1] = *(const u32x4*)(arp + k0 + 112);
    }
  }

#pragma unroll
  for (int g = 0; g < 8; ++g) {
    int col = n0 + g * 16 + l16;
    float badd = bias ? bias[col] : 0.f;
#pragma unroll
    for (int e = 0; e < 8; ++e) {
      long idx = (long)(m0 + half * 8 + e) * Nc + col;
      float val = acc[g][e] + badd;
      if (accumulate) C[idx] += val; else C[idx] = val;
    }
  }
}

// --------------------------------------------- per-degree gather + concat (bf16 out)
__global__ void k_gather_concat(const float* __restrict__ atom, const float* __restrict__ bond,
                                const int* __restrict__ anbr, const int* __restrict__ bnbr,
                                bf16* __restrict__ out, int fin, int deg) {
  long i = (long)blockIdx.x * blockDim.x + threadIdx.x;
  int width = fin + EDGE_F;
  long total = (long)ND_G * width;
  if (i >= total) return;
  int r = (int)(i / width);
  int c = (int)(i % width);
  float s = 0.f;
  if (c < fin) {
    const int* ar = anbr + (long)r * (deg + 1);
    for (int j = 0; j <= deg; ++j) s += atom[(long)ar[j] * fin + c];
  } else {
    int cb = c - fin;
    const int* br = bnbr + (long)r * deg;
    for (int j = 0; j < deg; ++j) s += bond[(long)br[j] * EDGE_F + cb];
  }
  out[i] = (bf16)s;
}

// ---------------------------------------------------------- BatchNorm (no affine)
__global__ void k_bn_stats(const float* __restrict__ act, float* __restrict__ stats, int fout) {
  int tid = blockIdx.x * blockDim.x + threadIdx.x;   // fout * BN_GROUPS threads
  int c = tid % fout;
  int g = tid / fout;
  float s = 0.f, sq = 0.f;
  for (int r = g; r < N_ATOMS; r += BN_GROUPS) {
    float v = act[(long)r * fout + c];
    s += v; sq += v * v;
  }
  atomicAdd(&stats[c], s);
  atomicAdd(&stats[fout + c], sq);
}

__global__ void k_bn_norm_relu(float* __restrict__ act, const float* __restrict__ stats, int fout) {
  long i = (long)blockIdx.x * blockDim.x + threadIdx.x;
  long total = (long)N_ATOMS * fout;
  if (i >= total) return;
  int c = (int)(i % fout);
  const float invN = 1.f / (float)N_ATOMS;
  float mu  = stats[c] * invN;
  float var = stats[fout + c] * invN - mu * mu;
  float v = (act[i] - mu) * rsqrtf(var + BN_EPS);
  act[i] = v > 0.f ? v : 0.f;
}

// -------------------------------------- row softmax (512 cols) + molecule segment-sum
__global__ __launch_bounds__(256)
void k_softmax_segsum(const float* __restrict__ Z, const int* __restrict__ mol_id,
                      float* __restrict__ fp) {
  int wave = threadIdx.x >> 5;
  int lane = threadIdx.x & 31;
  int row = blockIdx.x * 8 + wave;
  const float* z = Z + (long)row * 512;
  float v[16];
  float m = -3.402823466e38f;
#pragma unroll
  for (int j = 0; j < 16; ++j) { v[j] = z[lane + 32 * j]; m = fmaxf(m, v[j]); }
#pragma unroll
  for (int off = 16; off > 0; off >>= 1) m = fmaxf(m, __shfl_xor(m, off, 32));
  float s = 0.f;
#pragma unroll
  for (int j = 0; j < 16; ++j) { v[j] = __expf(v[j] - m); s += v[j]; }
#pragma unroll
  for (int off = 16; off > 0; off >>= 1) s += __shfl_xor(s, off, 32);
  float inv = 1.f / s;
  float* dst = fp + (long)mol_id[row] * 512;
#pragma unroll
  for (int j = 0; j < 16; ++j) atomicAdd(&dst[lane + 32 * j], v[j] * inv);
}

// ---------------------------------------------------------------- orchestration

extern "C" void kernel_launch(void* const* d_in, const int* in_sizes, int n_in,
                              void* d_out, int out_size, void* d_ws, size_t ws_size,
                              hipStream_t stream) {
  const float* atom = (const float*)d_in[0];
  const float* bond = (const float*)d_in[1];
  const int*   mol  = (const int*)d_in[2];
  const int* anbr[4] = { (const int*)d_in[3], (const int*)d_in[5],
                         (const int*)d_in[7], (const int*)d_in[9] };
  const int* bnbr[4] = { (const int*)d_in[4], (const int*)d_in[6],
                         (const int*)d_in[8], (const int*)d_in[10] };
  const float* Wself[2] = { (const float*)d_in[11], (const float*)d_in[17] };
  const float* biasl[2] = { (const float*)d_in[12], (const float*)d_in[18] };
  const float* Wdeg0[4] = { (const float*)d_in[13], (const float*)d_in[14],
                            (const float*)d_in[15], (const float*)d_in[16] };
  const float* Wdeg1[4] = { (const float*)d_in[19], (const float*)d_in[20],
                            (const float*)d_in[21], (const float*)d_in[22] };
  const float* Wout[3] = { (const float*)d_in[23], (const float*)d_in[25], (const float*)d_in[27] };
  const float* bout[3] = { (const float*)d_in[24], (const float*)d_in[26], (const float*)d_in[28] };
  float* fp = (float*)d_out;

  uint8_t* ws = (uint8_t*)d_ws;
  size_t off = 0;
  auto alloc = [&](size_t bytes) -> void* {
    void* p = ws + off;
    off = (off + bytes + 255) & ~(size_t)255;
    return p;
  };
  bf16* WT_self0 = (bf16*)alloc((size_t)256 * 128 * 2);
  bf16* WT_deg0[4]; for (int i = 0; i < 4; ++i) WT_deg0[i] = (bf16*)alloc((size_t)256 * 192 * 2);
  bf16* WT_self1 = (bf16*)alloc((size_t)512 * 256 * 2);
  bf16* WT_deg1[4]; for (int i = 0; i < 4; ++i) WT_deg1[i] = (bf16*)alloc((size_t)512 * 320 * 2);
  bf16* WT_out0 = (bf16*)alloc((size_t)512 * 128 * 2);
  bf16* WT_out1 = (bf16*)alloc((size_t)512 * 256 * 2);
  bf16* WT_out2 = (bf16*)alloc((size_t)512 * 512 * 2);
  bf16*  X16   = (bf16*)alloc((size_t)N_ATOMS * 512 * 2);   // bf16 activations (reused)
  bf16*  G16   = (bf16*)alloc((size_t)ND_G * 320 * 2);      // concat gather buffer (reused)
  float* Z     = (float*)alloc((size_t)N_ATOMS * 512 * 4);  // fp-update GEMM out (reused)
  float* act0  = (float*)alloc((size_t)N_ATOMS * 256 * 4);
  float* act1  = (float*)alloc((size_t)N_ATOMS * 512 * 4);
  float* stats = (float*)alloc(1024 * 4);

  auto cdiv = [](long a, long b) { return (unsigned)((a + b - 1) / b); };
  auto xpose = [&](const float* src, bf16* dst, int K, int Nc) {
    long t = (long)K * Nc;
    k_transpose_bf16<<<cdiv(t, 256), 256, 0, stream>>>(src, dst, K, Nc);
  };
  auto gemm = [&](const bf16* A, const bf16* WT, const float* bias, float* C,
                  int M, int K, int Nc, int accum) {
    dim3 grid(Nc / 128, M / 128);
    size_t shmem = (size_t)128 * (K + 8) * 2;
    int use_tdm = (K == 128 || K == 256 || K == 512) ? 1 : 0;
    k_gemm_bf16_wmma<<<grid, 256, shmem, stream>>>(A, WT, bias, C, M, K, Nc, accum, use_tdm);
  };

  // --- weight prep (tiny; all transposed + converted to bf16) ---
  xpose(Wself[0], WT_self0, 128, 256);
  for (int i = 0; i < 4; ++i) xpose(Wdeg0[i], WT_deg0[i], 192, 256);
  xpose(Wself[1], WT_self1, 256, 512);
  for (int i = 0; i < 4; ++i) xpose(Wdeg1[i], WT_deg1[i], 320, 512);
  xpose(Wout[0], WT_out0, 128, 512);
  xpose(Wout[1], WT_out1, 256, 512);
  xpose(Wout[2], WT_out2, 512, 512);

  // --- init fingerprint accumulator ---
  k_zero_f32<<<cdiv(out_size, 256), 256, 0, stream>>>(fp, out_size);

  // ===== stage 0: fp += softmax(atom@Wout0+b), act0 = conv0(atom) =====
  long n0e = (long)N_ATOMS * 128;
  k_f32_to_bf16<<<cdiv(n0e, 256), 256, 0, stream>>>(atom, X16, n0e);
  gemm(X16, WT_out0, bout[0], Z, N_ATOMS, 128, 512, 0);
  k_softmax_segsum<<<N_ATOMS / 8, 256, 0, stream>>>(Z, mol, fp);
  gemm(X16, WT_self0, biasl[0], act0, N_ATOMS, 128, 256, 0);
  for (int i = 0; i < 4; ++i) {
    int d = i + 1;
    long t = (long)ND_G * (128 + EDGE_F);
    k_gather_concat<<<cdiv(t, 256), 256, 0, stream>>>(atom, bond, anbr[i], bnbr[i], G16, 128, d);
    // self indices of degree group i are rows [i*ND, (i+1)*ND) -> accumulate in place
    gemm(G16, WT_deg0[i], nullptr, act0 + (size_t)i * ND_G * 256, ND_G, 192, 256, 1);
  }
  k_zero_f32<<<cdiv(1024, 256), 256, 0, stream>>>(stats, 1024);
  k_bn_stats<<<cdiv((long)256 * BN_GROUPS, 256), 256, 0, stream>>>(act0, stats, 256);
  k_bn_norm_relu<<<cdiv((long)N_ATOMS * 256, 256), 256, 0, stream>>>(act0, stats, 256);

  // ===== stage 1: fp += softmax(act0@Wout1+b), act1 = conv1(act0) =====
  long n1e = (long)N_ATOMS * 256;
  k_f32_to_bf16<<<cdiv(n1e, 256), 256, 0, stream>>>(act0, X16, n1e);
  gemm(X16, WT_out1, bout[1], Z, N_ATOMS, 256, 512, 0);
  k_softmax_segsum<<<N_ATOMS / 8, 256, 0, stream>>>(Z, mol, fp);
  gemm(X16, WT_self1, biasl[1], act1, N_ATOMS, 256, 512, 0);
  for (int i = 0; i < 4; ++i) {
    int d = i + 1;
    long t = (long)ND_G * (256 + EDGE_F);
    k_gather_concat<<<cdiv(t, 256), 256, 0, stream>>>(act0, bond, anbr[i], bnbr[i], G16, 256, d);
    gemm(G16, WT_deg1[i], nullptr, act1 + (size_t)i * ND_G * 512, ND_G, 320, 512, 1);
  }
  k_zero_f32<<<cdiv(1024, 256), 256, 0, stream>>>(stats, 1024);
  k_bn_stats<<<cdiv((long)512 * BN_GROUPS, 256), 256, 0, stream>>>(act1, stats, 512);
  k_bn_norm_relu<<<cdiv((long)N_ATOMS * 512, 256), 256, 0, stream>>>(act1, stats, 512);

  // ===== stage 2: fp += softmax(act1@Wout2+b) =====
  long n2e = (long)N_ATOMS * 512;
  k_f32_to_bf16<<<cdiv(n2e, 256), 256, 0, stream>>>(act1, X16, n2e);
  gemm(X16, WT_out2, bout[2], Z, N_ATOMS, 512, 512, 0);
  k_softmax_segsum<<<N_ATOMS / 8, 256, 0, stream>>>(Z, mol, fp);
}
